// AttentionClassifier_36137854828970
// MI455X (gfx1250) — matrix operand order repaired and checked by reading, not compiled, not run on hardware
//
#include <hip/hip_runtime.h>
#include <hip/hip_bf16.h>

#define BB   32
#define SS   512
#define DD   768
#define HH   12
#define DHD  64
#define IN3  3072
#define TT   (BB * SS)          // 16384 tokens

typedef __attribute__((ext_vector_type(16))) __bf16 bf16x16;
typedef __attribute__((ext_vector_type(8)))  __bf16 bf16x8;
typedef __attribute__((ext_vector_type(8)))  float  f32x8;

union BF16Frag { bf16x16 v; bf16x8 h[2]; __bf16 e[16]; };
union F32Acc   { f32x8   v; float  e[8]; };

// ---------------------------------------------------------------------------
// WMMA helpers (gfx1250, wave32). Layouts per CDNA5 ISA 7.12.2.
// ---------------------------------------------------------------------------

// A fragment: 16x32 bf16 tile of row-major A (lda in elements).
// lane l: m = l&15; K-halves [kb, kb+8) and [16+kb, 16+kb+8), kb = (l>>4)*8.
__device__ __forceinline__ bf16x16 load_frag_a(const __bf16* A, int lda, int row0, int k0) {
    int lane = threadIdx.x & 31;
    int m    = lane & 15;
    int kb   = (lane >> 4) * 8;
    const __bf16* p = A + (long)(row0 + m) * lda + k0 + kb;
    BF16Frag f;
    f.h[0] = *(const bf16x8*)(p);        // K = k0+kb .. +7
    f.h[1] = *(const bf16x8*)(p + 16);   // K = k0+16+kb .. +7
    return f.v;
}

// B fragment: 32x16 bf16 tile, B supplied TRANSPOSED (Bt is N x K row-major).
// lane l: n = l&15; K-halves [kb, kb+8) and [16+kb, 16+kb+8), kb = (l>>4)*8.
__device__ __forceinline__ bf16x16 load_frag_b(const __bf16* Bt, int ldb, int col0, int k0) {
    int lane = threadIdx.x & 31;
    int n    = lane & 15;
    int kb   = (lane >> 4) * 8;
    const __bf16* p = Bt + (long)(col0 + n) * ldb + k0 + kb;
    BF16Frag f;
    f.h[0] = *(const bf16x8*)(p);
    f.h[1] = *(const bf16x8*)(p + 16);
    return f.v;
}

__device__ __forceinline__ f32x8 wmma_bf16(bf16x16 a, bf16x16 b, f32x8 c) {
    return __builtin_amdgcn_wmma_f32_16x16x32_bf16(
        /*neg_a=*/false, a, /*neg_b=*/false, b,
        /*c_mod=*/(short)0, c, /*reuse_a=*/false, /*reuse_b=*/false);
}

// C/D 16x16 f32 tile store: lane l holds col = l&15, rows r + 8*(l>>4).
// mode 0: +bias; mode 1: +bias then exact GELU.
__device__ __forceinline__ void store_tile(__bf16* C, int ldc, int row0, int col0,
                                           f32x8 acc, const float* bias, int mode) {
    int lane  = threadIdx.x & 31;
    int n     = lane & 15;
    int rbase = (lane >> 4) * 8;
    float bv  = bias ? bias[col0 + n] : 0.0f;
    F32Acc a; a.v = acc;
#pragma unroll
    for (int r = 0; r < 8; ++r) {
        float v = a.e[r] + bv;
        if (mode == 1) v = 0.5f * v * (1.0f + erff(v * 0.70710678118654752f));
        C[(long)(row0 + rbase + r) * ldc + col0 + n] = (__bf16)v;
    }
}

// ---------------------------------------------------------------------------
// 1. Embedding gather + LayerNorm -> bf16 activations. One wave per token.
// ---------------------------------------------------------------------------
__global__ __launch_bounds__(256) void embed_ln(const int* __restrict__ ids,
                                                const float* __restrict__ tok,
                                                const float* __restrict__ pos,
                                                const float* __restrict__ g,
                                                const float* __restrict__ bt,
                                                __bf16* __restrict__ X) {
    int wave = (blockIdx.x * 256 + threadIdx.x) >> 5;
    int lane = threadIdx.x & 31;
    if (wave >= TT) return;
    int s  = wave & (SS - 1);
    int id = ids[wave];
    const float* tr = tok + (long)id * DD;
    const float* pr = pos + (long)s  * DD;

    float v[DD / 32];
    float sum = 0.0f;
#pragma unroll
    for (int i = 0; i < DD / 32; ++i) {
        float x = tr[lane + i * 32] + pr[lane + i * 32];
        v[i] = x; sum += x;
    }
#pragma unroll
    for (int m = 16; m; m >>= 1) sum += __shfl_xor(sum, m, 32);
    float mu = sum * (1.0f / DD);
    float var = 0.0f;
#pragma unroll
    for (int i = 0; i < DD / 32; ++i) { float d = v[i] - mu; var += d * d; }
#pragma unroll
    for (int m = 16; m; m >>= 1) var += __shfl_xor(var, m, 32);
    float inv = rsqrtf(var * (1.0f / DD) + 1e-5f);

    __bf16* xr = X + (long)wave * DD;
#pragma unroll
    for (int i = 0; i < DD / 32; ++i) {
        int d = lane + i * 32;
        xr[d] = (__bf16)((v[i] - mu) * inv * g[d] + bt[d]);
    }
}

// ---------------------------------------------------------------------------
// 2a. Pack per-head weights (H, D, DH) fp32 -> transposed bf16 (H*DH, D).
//     wT[col*D + d] = w[h, d, e], col = h*DH + e.
// ---------------------------------------------------------------------------
__global__ __launch_bounds__(256) void pack_headT(const float* __restrict__ w,
                                                  __bf16* __restrict__ wT) {
    int idx = blockIdx.x * 256 + threadIdx.x;
    if (idx >= DD * DD) return;
    int col = idx / DD, d = idx - col * DD;
    int h = col >> 6, e = col & 63;
    wT[idx] = (__bf16)w[((long)h * DD + d) * DHD + e];
}

// 2b. Transpose (D, N) fp32 -> (N, D) bf16.
__global__ __launch_bounds__(256) void transposeT(const float* __restrict__ w,
                                                  __bf16* __restrict__ wT, int N) {
    int idx = blockIdx.x * 256 + threadIdx.x;
    if (idx >= N * DD) return;
    int j = idx / DD, d = idx - j * DD;
    wT[idx] = (__bf16)w[(long)d * N + j];
}

// ---------------------------------------------------------------------------
// 3. Fused K/V projection GEMM: (16384x768)x(768x768) twice, shared A frags.
//    Each wave: 32x32 output tile for BOTH K and V; 8 WMMAs per K-step.
// ---------------------------------------------------------------------------
__global__ __launch_bounds__(256) void kv_proj(const __bf16* __restrict__ X,
                                               const __bf16* __restrict__ WkT,
                                               const __bf16* __restrict__ WvT,
                                               const float*  __restrict__ bk,
                                               const float*  __restrict__ bv,
                                               __bf16* __restrict__ Kb,
                                               __bf16* __restrict__ Vb) {
    int wave = (blockIdx.x * 256 + threadIdx.x) >> 5;
    const int NG = DD / 32;                 // 24 column groups
    int mg = wave / NG, ng = wave - mg * NG;
    if (mg * 32 >= TT) return;              // wave-uniform guard
    int row0 = mg * 32, col0 = ng * 32;

    f32x8 k00 = {}, k01 = {}, k10 = {}, k11 = {};
    f32x8 v00 = {}, v01 = {}, v10 = {}, v11 = {};

    for (int k0 = 0; k0 < DD; k0 += 32) {
        bf16x16 a0  = load_frag_a(X, DD, row0,      k0);
        bf16x16 a1  = load_frag_a(X, DD, row0 + 16, k0);
        bf16x16 kb0 = load_frag_b(WkT, DD, col0,      k0);
        bf16x16 kb1 = load_frag_b(WkT, DD, col0 + 16, k0);
        bf16x16 vb0 = load_frag_b(WvT, DD, col0,      k0);
        bf16x16 vb1 = load_frag_b(WvT, DD, col0 + 16, k0);
        k00 = wmma_bf16(a0, kb0, k00);
        k01 = wmma_bf16(a0, kb1, k01);
        k10 = wmma_bf16(a1, kb0, k10);
        k11 = wmma_bf16(a1, kb1, k11);
        v00 = wmma_bf16(a0, vb0, v00);
        v01 = wmma_bf16(a0, vb1, v01);
        v10 = wmma_bf16(a1, vb0, v10);
        v11 = wmma_bf16(a1, vb1, v11);
    }
    store_tile(Kb, DD, row0,      col0,      k00, bk, 0);
    store_tile(Kb, DD, row0,      col0 + 16, k01, bk, 0);
    store_tile(Kb, DD, row0 + 16, col0,      k10, bk, 0);
    store_tile(Kb, DD, row0 + 16, col0 + 16, k11, bk, 0);
    store_tile(Vb, DD, row0,      col0,      v00, bv, 0);
    store_tile(Vb, DD, row0,      col0 + 16, v01, bv, 0);
    store_tile(Vb, DD, row0 + 16, col0,      v10, bv, 0);
    store_tile(Vb, DD, row0 + 16, col0 + 16, v11, bv, 0);
}

// ---------------------------------------------------------------------------
// 4. Attention for query position 0 only. One wave per (b, h).
// ---------------------------------------------------------------------------
__global__ __launch_bounds__(32) void attn_cls(const __bf16* __restrict__ X,
                                               const __bf16* __restrict__ WqT,
                                               const float*  __restrict__ bq,
                                               const __bf16* __restrict__ Kb,
                                               const __bf16* __restrict__ Vb,
                                               __bf16* __restrict__ clsA) {
    __shared__ float q0[DHD];
    __shared__ float wts[SS];
    int bh = blockIdx.x;
    int b = bh / HH, h = bh - b * HH;
    int lane = threadIdx.x;
    const __bf16* x0 = X + (long)b * SS * DD;   // CLS token row

    // q0[e] = bq + x0 . WqT[h*64+e, :]
#pragma unroll
    for (int p = 0; p < 2; ++p) {
        int e = lane * 2 + p;
        const __bf16* wr = WqT + (long)(h * DHD + e) * DD;
        float acc = bq[h * DHD + e];
        for (int d = 0; d < DD; ++d) acc += (float)x0[d] * (float)wr[d];
        q0[e] = acc;
    }
    __syncthreads();

    // scores over 512 keys, 16 per lane
    float sc[SS / 32];
    float lmax = -1e30f;
#pragma unroll
    for (int i = 0; i < SS / 32; ++i) {
        int t = lane + i * 32;
        const __bf16* kr = Kb + (long)(b * SS + t) * DD + h * DHD;
        float s = 0.0f;
#pragma unroll
        for (int e = 0; e < DHD; ++e) s += q0[e] * (float)kr[e];
        s *= 0.125f;                    // 1/sqrt(64)
        sc[i] = s;
        lmax = fmaxf(lmax, s);
    }
#pragma unroll
    for (int m = 16; m; m >>= 1) lmax = fmaxf(lmax, __shfl_xor(lmax, m, 32));
    float lsum = 0.0f;
#pragma unroll
    for (int i = 0; i < SS / 32; ++i) {
        float e = expf(sc[i] - lmax);
        wts[lane + i * 32] = e;
        lsum += e;
    }
#pragma unroll
    for (int m = 16; m; m >>= 1) lsum += __shfl_xor(lsum, m, 32);
    float inv = 1.0f / lsum;
    __syncthreads();

    // out[e] = sum_t w[t] * V[t, e]
#pragma unroll
    for (int p = 0; p < 2; ++p) {
        int e = lane * 2 + p;
        float o = 0.0f;
        for (int t = 0; t < SS; ++t)
            o += wts[t] * (float)Vb[(long)(b * SS + t) * DD + h * DHD + e];
        clsA[b * DD + h * DHD + e] = (__bf16)(o * inv);
    }
}

// ---------------------------------------------------------------------------
// 5. Small WMMA GEMM for head: C = epi(A(MxK) * Bt(NxK)^T + bias).
//    One wave per 16x16 output tile.
// ---------------------------------------------------------------------------
__global__ __launch_bounds__(256) void gemm_head(const __bf16* __restrict__ A,
                                                 const __bf16* __restrict__ Bt,
                                                 const float*  __restrict__ bias,
                                                 __bf16* __restrict__ C,
                                                 int M, int N, int Kd, int mode) {
    int wave   = (blockIdx.x * 256 + threadIdx.x) >> 5;
    int ntiles = N >> 4;
    int mt = wave / ntiles, nt = wave - mt * ntiles;
    if (mt * 16 >= M) return;               // wave-uniform
    f32x8 acc = {};
    for (int k0 = 0; k0 < Kd; k0 += 32) {
        bf16x16 a = load_frag_a(A,  Kd, mt * 16, k0);
        bf16x16 b = load_frag_b(Bt, Kd, nt * 16, k0);
        acc = wmma_bf16(a, b, acc);
    }
    store_tile(C, N, mt * 16, nt * 16, acc, bias, mode);
}

// ---------------------------------------------------------------------------
// 6. Final logits: out[b] = sigmoid(h[b,:] . w2 + b2). One block of 256.
// ---------------------------------------------------------------------------
__global__ __launch_bounds__(256) void head_logits(const __bf16* __restrict__ hb,
                                                   const float* __restrict__ w2,
                                                   const float* __restrict__ b2,
                                                   float* __restrict__ out) {
    __shared__ float red[256];
    int t = threadIdx.x;
    int b = t >> 3, j = t & 7;
    float s = 0.0f;
    for (int i = j; i < IN3; i += 8) s += (float)hb[(long)b * IN3 + i] * w2[i];
    red[t] = s;
    __syncthreads();
    if (j == 0) {
        float tot = b2[0];
#pragma unroll
        for (int q = 0; q < 8; ++q) tot += red[(b << 3) + q];
        out[b] = 1.0f / (1.0f + expf(-tot));
    }
}

// ---------------------------------------------------------------------------
extern "C" void kernel_launch(void* const* d_in, const int* in_sizes, int n_in,
                              void* d_out, int out_size, void* d_ws, size_t ws_size,
                              hipStream_t stream) {
    const int*   ids   = (const int*)  d_in[0];
    const float* tok   = (const float*)d_in[1];
    const float* pos   = (const float*)d_in[2];
    const float* ln_g  = (const float*)d_in[3];
    const float* ln_b  = (const float*)d_in[4];
    const float* wq    = (const float*)d_in[5];
    const float* bq    = (const float*)d_in[6];
    const float* wk    = (const float*)d_in[7];
    const float* bk    = (const float*)d_in[8];
    const float* wv    = (const float*)d_in[9];
    const float* bv    = (const float*)d_in[10];
    const float* wo    = (const float*)d_in[11];
    const float* bo    = (const float*)d_in[12];
    const float* w1    = (const float*)d_in[13];
    const float* b1    = (const float*)d_in[14];
    const float* w2    = (const float*)d_in[15];
    const float* b2    = (const float*)d_in[16];
    float* outp        = (float*)d_out;

    // workspace carve-out (256B aligned)
    char*  ws  = (char*)d_ws;
    size_t off = 0;
    auto carve = [&](size_t bytes) -> void* {
        void* p = ws + off;
        off = (off + bytes + 255) & ~(size_t)255;
        return p;
    };
    __bf16* Xb   = (__bf16*)carve((size_t)TT * DD * 2);     // 24 MB LN output
    __bf16* WqT  = (__bf16*)carve((size_t)DD * DD * 2);
    __bf16* WkT  = (__bf16*)carve((size_t)DD * DD * 2);
    __bf16* WvT  = (__bf16*)carve((size_t)DD * DD * 2);
    __bf16* WoT  = (__bf16*)carve((size_t)DD * DD * 2);
    __bf16* W1T  = (__bf16*)carve((size_t)IN3 * DD * 2);
    __bf16* Kb   = (__bf16*)carve((size_t)TT * DD * 2);     // 24 MB
    __bf16* Vb   = (__bf16*)carve((size_t)TT * DD * 2);     // 24 MB
    __bf16* clsA = (__bf16*)carve((size_t)BB * DD * 2);
    __bf16* clsC = (__bf16*)carve((size_t)BB * DD * 2);
    __bf16* Hb   = (__bf16*)carve((size_t)BB * IN3 * 2);

    // weight conversion / transposition
    pack_headT<<<(DD * DD + 255) / 256, 256, 0, stream>>>(wq, WqT);
    pack_headT<<<(DD * DD + 255) / 256, 256, 0, stream>>>(wk, WkT);
    pack_headT<<<(DD * DD + 255) / 256, 256, 0, stream>>>(wv, WvT);
    transposeT<<<(DD  * DD + 255) / 256, 256, 0, stream>>>(wo, WoT, DD);
    transposeT<<<(IN3 * DD + 255) / 256, 256, 0, stream>>>(w1, W1T, IN3);

    // embedding + LayerNorm (16384 waves, 8 per block)
    embed_ln<<<TT / 8, 256, 0, stream>>>(ids, tok, pos, ln_g, ln_b, Xb);

    // K/V projection: (16384/32)*(768/32) = 12288 waves -> 1536 blocks
    kv_proj<<<(TT / 32) * (DD / 32) / 8, 256, 0, stream>>>(Xb, WkT, WvT, bk, bv, Kb, Vb);

    // attention at CLS (query 0 only): one wave per (b,h)
    attn_cls<<<BB * HH, 32, 0, stream>>>(Xb, WqT, bq, Kb, Vb, clsA);

    // cls_ctx = clsA @ wo + bo   : waves = (32/16)*(768/16) = 96 -> 12 blocks
    gemm_head<<<12, 256, 0, stream>>>(clsA, WoT, bo, clsC, BB, DD, DD, 0);

    // h = gelu(cls_ctx @ w1 + b1): waves = (32/16)*(3072/16) = 384 -> 48 blocks
    gemm_head<<<48, 256, 0, stream>>>(clsC, W1T, b1, Hb, BB, IN3, DD, 1);

    // out = sigmoid(h @ w2 + b2)
    head_logits<<<1, 256, 0, stream>>>(Hb, w2, b2, outp);
}
